// SelfAttention2D_2482491097326
// MI455X (gfx1250) — matrix operand order repaired and checked by reading, not compiled, not run on hardware
//
#include <hip/hip_runtime.h>
#include <hip/hip_bf16.h>
#include <math.h>

// ---------------------------------------------------------------------------
// SelfAttention2D for MI455X (gfx1250, wave32, WMMA).  B=8, C=64, H=W=256.
//   1) qkv_kernel : x fp32 -> q,k,v bf16 (+bias), pixels-as-M orientation so
//                   every lane stores 8 consecutive pixels with one b128 store
//   2) attn_kernel: per (b,c) 256x256x256 GEMM attn = q^T @ k (bf16->f32);
//                   transpose of q handled by LDS-gathered A fragments
//   3) stats_kernel: channel-softmax stats M=max_c, invD=1/sum_c exp
//   4) out_kernel : per (b,c) GEMM out = softmax(attn) @ v, exp(a-M)*invD
//                   applied on the fly while staging A into LDS as bf16
// Bandwidth-bound (AI ~24 FLOP/B): bf16 intermediates + fused normalization.
// ---------------------------------------------------------------------------

typedef __attribute__((ext_vector_type(16))) __bf16 v16bf;
typedef __attribute__((ext_vector_type(8)))  __bf16 v8bf;
typedef __attribute__((ext_vector_type(8)))  float  v8f;

#define BB   8
#define CC   64
#define WW   256
#define HWP  65536   // H*W

static __device__ __forceinline__ __bf16 f2bf(float f) { return (__bf16)f; }

// ---------------------------------------------------------------------------
// Kernel 1: fused QKV 1x1-conv.  Block = 256 thr, 128 pixels of one batch,
// all 64 output channels of q,k,v.  A = X^T (pixels x in-ch), B = W^T
// (in-ch x out-ch), D rows = pixels -> vectorized bf16 stores.
// ---------------------------------------------------------------------------
static __device__ __forceinline__ void
conv_out_tiles(const __bf16 (* __restrict__ sWm)[64],
               const float* __restrict__ sBm,
               const __bf16 (* __restrict__ sX)[136],
               __bf16* __restrict__ dst,      // already + b*CC*HWP
               int p0, int wave, int lr, int hi)
{
    // 8 pixel-tiles x 4 out-channel-tiles = 32 tiles round-robin over 8 waves
    for (int t = wave; t < 32; t += 8) {
        const int mt = t & 7;    // pixel tile (16 px)
        const int nt = t >> 3;   // out-channel tile (16 ch)
        v8f acc = {};
        #pragma unroll
        for (int k0 = 0; k0 < 64; k0 += 32) {
            v16bf a, bm;
            const int apix = mt * 16 + lr;     // A row = pixel
            const int ak   = k0 + hi * 8;      // A lane K: {0..7,16..23}/{8..15,24..31}
            #pragma unroll
            for (int e = 0; e < 8; ++e) {
                a[e]     = sX[ak + e][apix];
                a[8 + e] = sX[ak + 16 + e][apix];
            }
            const int orow = nt * 16 + lr;     // B col = out channel
            const int bk0  = k0 + hi * 8;      // B lane K: {0..7,16..23}/{8..15,24..31}
            #pragma unroll
            for (int e = 0; e < 16; ++e) bm[e] = sWm[orow][bk0 + e + (e & 8)];
            acc = __builtin_amdgcn_wmma_f32_16x16x32_bf16(
                false, a, false, bm, (short)0, acc, false, false);
        }
        // D: lane holds out-channel o = nt*16+lr, pixels (r + 8*hi) contiguous
        const int o = nt * 16 + lr;
        const float bias = sBm[o];
        v8bf pk;
        #pragma unroll
        for (int r = 0; r < 8; ++r) pk[r] = f2bf(acc[r] + bias);
        *(v8bf*)&dst[(size_t)o * HWP + p0 + mt * 16 + hi * 8] = pk;
    }
}

__global__ void __launch_bounds__(256)
qkv_kernel(const float* __restrict__ x,
           const float* __restrict__ Wq, const float* __restrict__ bq,
           const float* __restrict__ Wk, const float* __restrict__ bk,
           const float* __restrict__ Wv, const float* __restrict__ bv,
           __bf16* __restrict__ qbuf, __bf16* __restrict__ kbuf,
           __bf16* __restrict__ vbuf)
{
    __shared__ __bf16 sW[3][64][64];   // 24 KB (Wq/Wk/Wv as bf16)
    __shared__ float  sBias[3][64];
    __shared__ __bf16 sX[64][136];     // 17 KB (64 in-ch x 128 px, padded)

    const int tid = threadIdx.x;
    const int b   = blockIdx.x >> 9;           // 512 blocks per batch
    const int p0  = (blockIdx.x & 511) * 128;  // pixel tile start

    // W matrices: float4 loads, bf16 LDS stores (merge to ds_store_b64)
    for (int ch = tid; ch < 1024; ch += 256) {
        const int r = ch >> 4, c4 = (ch & 15) * 4;
        const float4 aq = *(const float4*)&Wq[r * 64 + c4];
        const float4 ak = *(const float4*)&Wk[r * 64 + c4];
        const float4 av = *(const float4*)&Wv[r * 64 + c4];
        sW[0][r][c4] = f2bf(aq.x); sW[0][r][c4+1] = f2bf(aq.y);
        sW[0][r][c4+2] = f2bf(aq.z); sW[0][r][c4+3] = f2bf(aq.w);
        sW[1][r][c4] = f2bf(ak.x); sW[1][r][c4+1] = f2bf(ak.y);
        sW[1][r][c4+2] = f2bf(ak.z); sW[1][r][c4+3] = f2bf(ak.w);
        sW[2][r][c4] = f2bf(av.x); sW[2][r][c4+1] = f2bf(av.y);
        sW[2][r][c4+2] = f2bf(av.z); sW[2][r][c4+3] = f2bf(av.w);
    }
    if (tid < 64) {
        sBias[0][tid] = bq[tid];
        sBias[1][tid] = bk[tid];
        sBias[2][tid] = bv[tid];
    }
    // x tile: float4 loads
    const size_t xbase = (size_t)b * CC * HWP + p0;
    for (int ch = tid; ch < 2048; ch += 256) {
        const int c = ch >> 5, p4 = (ch & 31) * 4;
        const float4 xv = *(const float4*)&x[xbase + (size_t)c * HWP + p4];
        sX[c][p4]     = f2bf(xv.x);
        sX[c][p4 + 1] = f2bf(xv.y);
        sX[c][p4 + 2] = f2bf(xv.z);
        sX[c][p4 + 3] = f2bf(xv.w);
    }
    __syncthreads();

    const int wave = tid >> 5;
    const int lane = tid & 31;
    const int lr   = lane & 15;
    const int hi   = lane >> 4;
    const size_t bofs = (size_t)b * CC * HWP;

    conv_out_tiles(sW[0], sBias[0], sX, qbuf + bofs, p0, wave, lr, hi);
    conv_out_tiles(sW[1], sBias[1], sX, kbuf + bofs, p0, wave, lr, hi);
    conv_out_tiles(sW[2], sBias[2], sX, vbuf + bofs, p0, wave, lr, hi);
}

// ---------------------------------------------------------------------------
// Kernel 2: attn[b,c] = q[b,c]^T @ k[b,c]  (256x256, K=256 over h), f32 out.
// Both strips staged as [K=h][col]; A fragments gathered (transposed) from
// the q strip.  Block tile 128x128, 8 waves in 4x2, each wave 32x64.
// ---------------------------------------------------------------------------
__global__ void __launch_bounds__(256)
attn_kernel(const __bf16* __restrict__ qbuf, const __bf16* __restrict__ kbuf,
            float* __restrict__ attn)
{
    __shared__ __bf16 sQ[32][144];   // 9 KB: 32 K-rows x 128 i-cols (pad 16)
    __shared__ __bf16 sK[32][144];   // 9 KB: 32 K-rows x 128 j-cols (pad 16)

    const int tid = threadIdx.x;
    const int bc  = blockIdx.x >> 2;
    const int ti  = ((blockIdx.x >> 1) & 1) * 128;
    const int tj  = (blockIdx.x & 1) * 128;
    const size_t base = (size_t)bc * HWP;

    const int wave = tid >> 5, lane = tid & 31;
    const int lr = lane & 15, hi = lane >> 4;
    const int wrow = wave >> 1, wcol = wave & 1;

    v8f acc[2][4] = {};

    for (int k0 = 0; k0 < 256; k0 += 32) {
        for (int ch = tid; ch < 512; ch += 256) {
            const int r = ch >> 4, c8 = (ch & 15) * 8;
            const size_t row = base + (size_t)(k0 + r) * WW;
            *(uint4*)&sQ[r][c8] = *(const uint4*)&qbuf[row + ti + c8];
            *(uint4*)&sK[r][c8] = *(const uint4*)&kbuf[row + tj + c8];
        }
        if (k0 + 32 < 256) {   // hint next K-strip (global_prefetch_b8)
            const size_t nrow = base + (size_t)(k0 + 32 + (tid >> 3)) * WW;
            __builtin_prefetch(&qbuf[nrow + ti + (tid & 7) * 16], 0, 0);
            __builtin_prefetch(&kbuf[nrow + tj + (tid & 7) * 16], 0, 0);
        }
        __syncthreads();

        v16bf a[2], bm[4];
        #pragma unroll
        for (int mi = 0; mi < 2; ++mi) {       // A = q^T: gather transposed
            const int row = wrow * 32 + mi * 16 + lr;
            const int ak  = hi * 8;
            #pragma unroll
            for (int e = 0; e < 8; ++e) {
                a[mi][e]     = sQ[ak + e][row];
                a[mi][8 + e] = sQ[ak + 16 + e][row];
            }
        }
        #pragma unroll
        for (int ni = 0; ni < 4; ++ni) {
            const int col = wcol * 64 + ni * 16 + lr;
            const int kk  = hi * 8;
            #pragma unroll
            for (int e = 0; e < 16; ++e) bm[ni][e] = sK[kk + e + (e & 8)][col];
        }
        #pragma unroll
        for (int mi = 0; mi < 2; ++mi)
            #pragma unroll
            for (int ni = 0; ni < 4; ++ni)
                acc[mi][ni] = __builtin_amdgcn_wmma_f32_16x16x32_bf16(
                    false, a[mi], false, bm[ni], (short)0, acc[mi][ni], false, false);
        __syncthreads();
    }

    #pragma unroll
    for (int mi = 0; mi < 2; ++mi)
        #pragma unroll
        for (int ni = 0; ni < 4; ++ni) {
            const int gi = ti + wrow * 32 + mi * 16 + hi * 8;
            const int gj = tj + wcol * 64 + ni * 16 + lr;
            #pragma unroll
            for (int r = 0; r < 8; ++r)
                attn[base + (size_t)(gi + r) * WW + gj] = acc[mi][ni][r];
        }
}

// ---------------------------------------------------------------------------
// Kernel 3: channel-softmax stats.  One thread per (b,i,j):
// M = max_c attn, invD = 1 / sum_c exp(attn - M).  attn read once, in regs.
// ---------------------------------------------------------------------------
__global__ void __launch_bounds__(256)
stats_kernel(const float* __restrict__ attn,
             float* __restrict__ Mx, float* __restrict__ invD)
{
    const size_t idx = (size_t)blockIdx.x * 256 + threadIdx.x;  // B*HW threads
    const int b  = (int)(idx >> 16);
    const int ij = (int)(idx & 65535);
    const float* p = attn + (size_t)b * CC * HWP + ij;

    float v[64];
    float m = -INFINITY;
    #pragma unroll
    for (int c = 0; c < 64; ++c) {
        v[c] = p[(size_t)c * HWP];
        m = fmaxf(m, v[c]);
    }
    float s = 0.f;
    #pragma unroll
    for (int c = 0; c < 64; ++c) s += __expf(v[c] - m);
    Mx[idx]   = m;
    invD[idx] = 1.0f / s;
}

// ---------------------------------------------------------------------------
// Kernel 4: out[b,c] = softmax(attn)[b,c] @ v[b,c].  A strip normalized on
// the fly (exp(a-M)*invD) into bf16 LDS with float4 global reads.
// ---------------------------------------------------------------------------
__global__ void __launch_bounds__(256)
out_kernel(const float* __restrict__ attn,
           const float* __restrict__ Mx, const float* __restrict__ invD,
           const __bf16* __restrict__ vbuf, float* __restrict__ out)
{
    __shared__ __bf16 sA[128][48];   // 12 KB: 128 i-rows x 32 m-cols (pad 16)
    __shared__ __bf16 sB[32][144];   //  9 KB: 32 m-rows x 128 j-cols (pad 16)

    const int tid = threadIdx.x;
    const int bc  = blockIdx.x >> 2;
    const int b   = bc >> 6;
    const int ti  = ((blockIdx.x >> 1) & 1) * 128;
    const int tj  = (blockIdx.x & 1) * 128;

    const size_t abase = (size_t)bc * HWP;
    const size_t sbase = (size_t)b * HWP;

    const int wave = tid >> 5, lane = tid & 31;
    const int lr = lane & 15, hi = lane >> 4;
    const int wrow = wave >> 1, wcol = wave & 1;

    v8f acc[2][4] = {};

    for (int m0 = 0; m0 < 256; m0 += 32) {
        // A strip: exp-normalize fp32 attn into bf16 LDS (float4 reads)
        for (int ch = tid; ch < 1024; ch += 256) {
            const int r = ch >> 3, c4 = (ch & 7) * 4;
            const int gi = ti + r, gm = m0 + c4;
            const float4 av = *(const float4*)&attn[abase + (size_t)gi * WW + gm];
            const float4 mv = *(const float4*)&Mx  [sbase + (size_t)gi * WW + gm];
            const float4 dv = *(const float4*)&invD[sbase + (size_t)gi * WW + gm];
            sA[r][c4]     = f2bf(__expf(av.x - mv.x) * dv.x);
            sA[r][c4 + 1] = f2bf(__expf(av.y - mv.y) * dv.y);
            sA[r][c4 + 2] = f2bf(__expf(av.z - mv.z) * dv.z);
            sA[r][c4 + 3] = f2bf(__expf(av.w - mv.w) * dv.w);
        }
        // B strip: v rows m0..m0+31
        for (int ch = tid; ch < 512; ch += 256) {
            const int r = ch >> 4, c8 = (ch & 15) * 8;
            *(uint4*)&sB[r][c8] =
                *(const uint4*)&vbuf[abase + (size_t)(m0 + r) * WW + tj + c8];
        }
        if (m0 + 32 < 256) {
            __builtin_prefetch(&attn[abase + (size_t)(ti + (tid & 127)) * WW + m0 + 32], 0, 0);
            __builtin_prefetch(&vbuf[abase + (size_t)(m0 + 32 + (tid >> 3)) * WW + tj + (tid & 7) * 16], 0, 0);
        }
        __syncthreads();

        v16bf a[2], bm[4];
        #pragma unroll
        for (int mi = 0; mi < 2; ++mi) {       // A rows contiguous in LDS
            const int row = wrow * 32 + mi * 16 + lr;
            const int ak  = hi * 8;
            #pragma unroll
            for (int e = 0; e < 8; ++e) {
                a[mi][e]     = sA[row][ak + e];
                a[mi][8 + e] = sA[row][ak + 16 + e];
            }
        }
        #pragma unroll
        for (int ni = 0; ni < 4; ++ni) {
            const int col = wcol * 64 + ni * 16 + lr;
            const int kk  = hi * 8;
            #pragma unroll
            for (int e = 0; e < 16; ++e) bm[ni][e] = sB[kk + e + (e & 8)][col];
        }
        #pragma unroll
        for (int mi = 0; mi < 2; ++mi)
            #pragma unroll
            for (int ni = 0; ni < 4; ++ni)
                acc[mi][ni] = __builtin_amdgcn_wmma_f32_16x16x32_bf16(
                    false, a[mi], false, bm[ni], (short)0, acc[mi][ni], false, false);
        __syncthreads();
    }

    #pragma unroll
    for (int mi = 0; mi < 2; ++mi)
        #pragma unroll
        for (int ni = 0; ni < 4; ++ni) {
            const int gi = ti + wrow * 32 + mi * 16 + hi * 8;
            const int gj = tj + wcol * 64 + ni * 16 + lr;
            #pragma unroll
            for (int r = 0; r < 8; ++r)
                out[abase + (size_t)(gi + r) * WW + gj] = acc[mi][ni][r];
        }
}

// ---------------------------------------------------------------------------
extern "C" void kernel_launch(void* const* d_in, const int* in_sizes, int n_in,
                              void* d_out, int out_size, void* d_ws, size_t ws_size,
                              hipStream_t stream)
{
    const float* x  = (const float*)d_in[0];
    const float* Wq = (const float*)d_in[1];
    const float* bq = (const float*)d_in[2];
    const float* Wk = (const float*)d_in[3];
    const float* bk = (const float*)d_in[4];
    const float* Wv = (const float*)d_in[5];
    const float* bv = (const float*)d_in[6];
    float* out = (float*)d_out;
    char*  ws  = (char*)d_ws;

    // Workspace layout (total ~324 MB):
    //   [0,64MB)      q    bf16  [b][c][h][w]
    //   [64,128MB)    k    bf16  [b][c][h][w]
    //   [128,192MB)   v    bf16  [b][c][h][w]
    //   [192,320MB)   attn fp32  [b][c][i][j]
    //   [320,322MB)   M    fp32  [b][i][j]
    //   [322,324MB)   invD fp32  [b][i][j]
    const size_t QKV_BYTES  = (size_t)BB * CC * HWP * 2;              // 64 MB
    const size_t ATTN_BYTES = (size_t)BB * CC * HWP * sizeof(float);  // 128 MB
    __bf16* qbuf = (__bf16*)(ws);
    __bf16* kbuf = (__bf16*)(ws + QKV_BYTES);
    __bf16* vbuf = (__bf16*)(ws + 2 * QKV_BYTES);
    float*  attn = (float*) (ws + 3 * QKV_BYTES);
    float*  Mx   = (float*) (ws + 3 * QKV_BYTES + ATTN_BYTES);
    float*  invD = (float*) (ws + 3 * QKV_BYTES + ATTN_BYTES
                                 + (size_t)BB * HWP * sizeof(float));

    qkv_kernel  <<<BB * (HWP / 128), 256, 0, stream>>>(x, Wq, bq, Wk, bk, Wv, bv,
                                                       qbuf, kbuf, vbuf);
    attn_kernel <<<BB * CC * 4,      256, 0, stream>>>(qbuf, kbuf, attn);
    stats_kernel<<<(BB * HWP) / 256, 256, 0, stream>>>(attn, Mx, invD);
    out_kernel  <<<BB * CC * 4,      256, 0, stream>>>(attn, Mx, invD, vbuf, out);

    (void)in_sizes; (void)n_in; (void)out_size; (void)ws_size;
}
